// ASRNN_general_12120397709832
// MI455X (gfx1250) — matrix operand order repaired and checked by reading, not compiled, not run on hardware
//
#include <hip/hip_runtime.h>
#include <hip/hip_bf16.h>

typedef _Float16 v16h __attribute__((ext_vector_type(16)));
typedef _Float16 v8h  __attribute__((ext_vector_type(8)));
typedef float    v8f  __attribute__((ext_vector_type(8)));

#define HID   512
#define NB    256
#define TT    784
#define NLAY  3

// ---- workspace layout (bytes) ----
#define OFF_W16   0                         // 4 * 512*512 halves (wh1, wi2, wh2, wi3) = 2 MB
#define OFF_XT    (4*512*512*2)             // xT[784][256] f32
#define OFF_PAR   (OFF_XT + TT*NB*4)        // 3 layers * 6 arrays * 512 * 4B
#define OFF_STATE (OFF_PAR + 3*6*512*4)     // 10 arrays * (512*256) f32
#define ST_SZ     (HID*NB)                  // 131072 elements per state array

// =====================================================================
// Prep kernels
// =====================================================================
__global__ void k_cvt_w(const float* __restrict__ w1, const float* __restrict__ w2,
                        const float* __restrict__ w3, const float* __restrict__ w4,
                        _Float16* __restrict__ dst) {
    int i = blockIdx.x * blockDim.x + threadIdx.x;          // 0 .. 262143
    if (i >= 512*512) return;
    dst[i]              = (_Float16)w1[i];
    dst[512*512   + i]  = (_Float16)w2[i];
    dst[2*512*512 + i]  = (_Float16)w3[i];
    dst[3*512*512 + i]  = (_Float16)w4[i];
}

__global__ void k_xt(const float* __restrict__ x, float* __restrict__ xT) {
    int i = blockIdx.x * blockDim.x + threadIdx.x;          // 0 .. 256*784-1
    if (i >= NB*TT) return;
    int n = i / TT, c = i % TT;
    xT[c*NB + n] = x[i];
}

// per-neuron params: [alpha][ro][cbias][cyc][shift][on] per layer, 512 each
__global__ void k_params(const float* ta1, const float* ta2, const float* ta3,
                         const float* tm1, const float* tm2, const float* tm3,
                         const float* bi1, const float* bh1,
                         const float* bi2, const float* bh2,
                         const float* bi3, float* __restrict__ par) {
    int i = blockIdx.x * blockDim.x + threadIdx.x;
    if (i >= NLAY*HID) return;
    int L = i / HID, h = i % HID;
    const float* ta = (L == 0) ? ta1 : (L == 1) ? ta2 : ta3;
    const float* tm = (L == 0) ? tm1 : (L == 1) ? tm2 : tm3;
    const double cmin[3] = {2.0, 4.0, 8.0};
    const double cmax[3] = {8.0, 16.0, 32.0};
    const double smax[3] = {4.0, 8.0, 16.0};  // float(int(0.5 * c_max))
    double f   = (double)h / 511.0;
    double dc  = 0.2 + 0.6 * f;
    int cyc = (int)rint(cmin[L] + (cmax[L] - cmin[L]) * f);   // round-half-even == np.round
    int on  = (int)rint(dc * (double)cyc);
    int ps  = (int)rint(smax[L] * f);
    int sh  = (ps > 0 && ps < cyc) ? ps : 0;                  // python slice semantics for ps>=cyc
    float cb = (L == 0) ? (bi1[h] + bh1[h]) : (L == 1) ? (bi2[h] + bh2[h]) : bi3[h];
    float* pl = par + L * 6 * HID;
    pl[h]        = (float)exp(-1.0 / (double)tm[h]);          // alpha
    pl[HID + h]  = (float)exp(-1.0 / (double)ta[h]);          // ro
    pl[2*HID + h] = cb;
    ((int*)pl)[3*HID + h] = cyc;
    ((int*)pl)[4*HID + h] = sh;
    ((int*)pl)[5*HID + h] = on;
}

__global__ void k_init(float* __restrict__ st) {
    int i = blockIdx.x * blockDim.x + threadIdx.x;
    if (i >= 10*ST_SZ) return;
    int a = i / ST_SZ;                  // 0-2 mem, 3-5 b, 6-8 ssum, 9 spk3
    st[i] = (a >= 3 && a < 6) ? 0.01f : 0.0f;
}

// =====================================================================
// Fused 4-tile WMMA k-loop: one shared A fragment (LDS spikes) feeds 4
// independent accumulator chains (4 n-tiles of the same weight matrix).
//   A layout (16-bit 16x32): lane L: m=L%16, halves k0+16*(L/16)+{0..15}.
//   B layout (16-bit 32x16): lane L: n=L%16, halves k0+16*(L/16)+{0..15}.
// =====================================================================
__device__ __forceinline__ void wmma_acc4(v8f acc[4],
                                          const _Float16* __restrict__ Wmat,  // matrix base
                                          int nbase,                          // wave's first tile col
                                          const _Float16* As,                 // LDS, row m stride HID
                                          int lane) {
    const int idx  = lane & 15;
    const int half = lane >> 4;
    const _Float16* arow = As + idx * HID + half * 16;
    const _Float16* brow = Wmat + (nbase + idx) * HID + half * 16;  // tiles offset by 16*HID
    #pragma unroll 2
    for (int k0 = 0; k0 < HID; k0 += 32) {
        union { v16h v; v8h h[2]; } A, B0, B1, B2, B3;
        A.h[0]  = *(const v8h*)(arow + k0);
        A.h[1]  = *(const v8h*)(arow + k0 + 8);
        B0.h[0] = *(const v8h*)(brow + k0);
        B0.h[1] = *(const v8h*)(brow + k0 + 8);
        B1.h[0] = *(const v8h*)(brow + 16*HID + k0);
        B1.h[1] = *(const v8h*)(brow + 16*HID + k0 + 8);
        B2.h[0] = *(const v8h*)(brow + 32*HID + k0);
        B2.h[1] = *(const v8h*)(brow + 32*HID + k0 + 8);
        B3.h[0] = *(const v8h*)(brow + 48*HID + k0);
        B3.h[1] = *(const v8h*)(brow + 48*HID + k0 + 8);
        acc[0] = __builtin_amdgcn_wmma_f32_16x16x32_f16(false, A.v, false, B0.v, (short)0, acc[0], false, false);
        acc[1] = __builtin_amdgcn_wmma_f32_16x16x32_f16(false, A.v, false, B1.v, (short)0, acc[1], false, false);
        acc[2] = __builtin_amdgcn_wmma_f32_16x16x32_f16(false, A.v, false, B2.v, (short)0, acc[2], false, false);
        acc[3] = __builtin_amdgcn_wmma_f32_16x16x32_f16(false, A.v, false, B3.v, (short)0, acc[3], false, false);
    }
}

__device__ __forceinline__ void zero4(v8f acc[4]) {
    #pragma unroll
    for (int j = 0; j < 4; ++j) {
        v8f z = {0.f,0.f,0.f,0.f,0.f,0.f,0.f,0.f};
        acc[j] = z;
    }
}

__device__ __forceinline__ float neuron_step(float inp, float& mem, float& b,
                                             float spk_old, float alpha, float ro, int mk) {
    float bn  = ro * b + (1.0f - ro) * spk_old;
    float Bth = 0.01f + 1.8f * bn;
    float mn  = mem * alpha + (1.0f - alpha) * inp - Bth * spk_old;  // DT = 1
    mn = mk ? mn : mem;
    float sn = (mk && (mn - Bth > 0.0f)) ? 1.0f : 0.0f;
    mem = mn; b = bn;
    return sn;
}

// =====================================================================
// Main persistent kernel: 16 workgroups (one per 16-sample batch group),
// 256 threads (8 waves x 4 n-tiles = 512 hidden units). No grid sync needed:
// the recurrence is batch-local. Whole-T loop inside the kernel.
// =====================================================================
__launch_bounds__(256, 1)
__global__ void k_main(const _Float16* __restrict__ W16, const float* __restrict__ xT,
                       const float* __restrict__ par, float* __restrict__ state,
                       const float* __restrict__ wi1) {
    __shared__ _Float16 s_spk1[16 * HID];
    __shared__ _Float16 s_spk2[16 * HID];

    const int g    = blockIdx.x;            // batch group: rows [16g, 16g+16)
    const int tid  = threadIdx.x;
    const int wave = tid >> 5;
    const int lane = tid & 31;
    const int idx  = lane & 15;
    const int mloc = (lane >> 4) * 8;       // local C-row base (8 rows per lane)
    const int mg   = g * 16 + mloc;         // global batch base for this lane
    const int nbase = wave * 64;            // wave's first tile column

    const _Float16* wh1 = W16;
    const _Float16* wi2 = W16 + 512*512;
    const _Float16* wh2 = W16 + 2*512*512;
    const _Float16* wi3 = W16 + 3*512*512;

    float* mem1 = state;               float* mem2 = state + 1*ST_SZ;  float* mem3 = state + 2*ST_SZ;
    float* bb1  = state + 3*ST_SZ;     float* bb2  = state + 4*ST_SZ;  float* bb3  = state + 5*ST_SZ;
    float* ss1  = state + 6*ST_SZ;     float* ss2  = state + 7*ST_SZ;  float* ss3  = state + 8*ST_SZ;
    float* spk3 = state + 9*ST_SZ;

    for (int i = tid; i < 16 * HID; i += 256) {
        s_spk1[i] = (_Float16)0.0f;
        s_spk2[i] = (_Float16)0.0f;
    }
    __syncthreads();

    int hcol[4];
    #pragma unroll
    for (int j = 0; j < 4; ++j) hcol[j] = nbase + j*16 + idx;

    for (int t = 0; t < TT; ++t) {
        v8f acc[4];
        // ---------------- Phase A: layer 1 (A = spk1(t-1)) ----------------
        zero4(acc);
        wmma_acc4(acc, wh1, nbase, s_spk1, lane);
        __syncthreads();                       // all reads of s_spk1 complete
        {
            const float* pl = par;             // layer 0 params
            const float* px = xT + t*NB + mg;  // 8 contiguous batch samples
            #pragma unroll
            for (int j = 0; j < 4; ++j) {
                int h = hcol[j];
                float alpha = pl[h], ro = pl[HID+h], cb = pl[2*HID+h];
                int cyc = ((const int*)pl)[3*HID+h], sh = ((const int*)pl)[4*HID+h],
                    on  = ((const int*)pl)[5*HID+h];
                int mk = ((((t % cyc) - sh) + cyc) % cyc) < on;
                float w1 = wi1[h];
                float* pm = mem1 + h*NB + mg;
                float* pb = bb1  + h*NB + mg;
                float* ps = ss1  + h*NB + mg;
                #pragma unroll
                for (int r = 0; r < 8; ++r) {
                    float so = (float)s_spk1[(mloc + r)*HID + h];
                    float inp = acc[j][r] + cb + px[r]*w1;
                    float m = pm[r], b = pb[r];
                    float sn = neuron_step(inp, m, b, so, alpha, ro, mk);
                    pm[r] = m; pb[r] = b; ps[r] += sn;
                    s_spk1[(mloc + r)*HID + h] = (_Float16)sn;   // own slot only
                }
            }
        }
        __syncthreads();                       // s_spk1 now holds spk1(t)

        // ---------------- Phase B: layer 2 (spk1(t)@wi2 + spk2(t-1)@wh2) ----
        zero4(acc);
        wmma_acc4(acc, wi2, nbase, s_spk1, lane);
        wmma_acc4(acc, wh2, nbase, s_spk2, lane);
        __syncthreads();                       // all reads of s_spk2 complete
        {
            const float* pl = par + 6*HID;     // layer 1 params
            #pragma unroll
            for (int j = 0; j < 4; ++j) {
                int h = hcol[j];
                float alpha = pl[h], ro = pl[HID+h], cb = pl[2*HID+h];
                int cyc = ((const int*)pl)[3*HID+h], sh = ((const int*)pl)[4*HID+h],
                    on  = ((const int*)pl)[5*HID+h];
                int mk = ((((t % cyc) - sh) + cyc) % cyc) < on;
                float* pm = mem2 + h*NB + mg;
                float* pb = bb2  + h*NB + mg;
                float* ps = ss2  + h*NB + mg;
                #pragma unroll
                for (int r = 0; r < 8; ++r) {
                    float so = (float)s_spk2[(mloc + r)*HID + h];
                    float inp = acc[j][r] + cb;
                    float m = pm[r], b = pb[r];
                    float sn = neuron_step(inp, m, b, so, alpha, ro, mk);
                    pm[r] = m; pb[r] = b; ps[r] += sn;
                    s_spk2[(mloc + r)*HID + h] = (_Float16)sn;
                }
            }
        }
        __syncthreads();                       // s_spk2 now holds spk2(t)

        // ---------------- Phase C: layer 3 (spk2(t)@wi3); no LDS writes ----
        zero4(acc);
        wmma_acc4(acc, wi3, nbase, s_spk2, lane);
        {
            const float* pl = par + 12*HID;    // layer 2 params
            #pragma unroll
            for (int j = 0; j < 4; ++j) {
                int h = hcol[j];
                float alpha = pl[h], ro = pl[HID+h], cb = pl[2*HID+h];
                int cyc = ((const int*)pl)[3*HID+h], sh = ((const int*)pl)[4*HID+h],
                    on  = ((const int*)pl)[5*HID+h];
                int mk = ((((t % cyc) - sh) + cyc) % cyc) < on;
                float* pm = mem3 + h*NB + mg;
                float* pb = bb3  + h*NB + mg;
                float* ps = ss3  + h*NB + mg;
                float* pk = spk3 + h*NB + mg;
                #pragma unroll
                for (int r = 0; r < 8; ++r) {
                    float so = pk[r];
                    float inp = acc[j][r] + cb;
                    float m = pm[r], b = pb[r];
                    float sn = neuron_step(inp, m, b, so, alpha, ro, mk);
                    pm[r] = m; pb[r] = b; ps[r] += sn; pk[r] = sn;
                }
            }
        }
        __syncthreads();   // orders C reads of s_spk2 / next-step A reads of s_spk1
    }
}

// =====================================================================
// Finalize: outputs = (sum_t spk3) @ wo.T / T + bo ; s*/T ; layer firing rates
// =====================================================================
__global__ void k_out(const float* __restrict__ ss3, const float* __restrict__ wo,
                      const float* __restrict__ bo, float* __restrict__ out) {
    int i = blockIdx.x * blockDim.x + threadIdx.x;   // 0..2559
    if (i >= NB * 10) return;
    int n = i / 10, o = i % 10;
    float s = 0.0f;
    for (int h = 0; h < HID; ++h) s += ss3[h*NB + n] * wo[o*HID + h];
    out[n*10 + o] = s / (float)TT + bo[o];
}

__global__ void k_srates(const float* __restrict__ ss, float* __restrict__ dst) {
    int i = blockIdx.x * blockDim.x + threadIdx.x;   // 0 .. 3*131072-1
    if (i >= NLAY * ST_SZ) return;
    int L = i / ST_SZ, r = i % ST_SZ;
    int h = r / NB, n = r % NB;
    dst[L*ST_SZ + n*HID + h] = ss[i] / (float)TT;    // transpose [h][n] -> [n][h]
}

__global__ void k_fr(const float* __restrict__ ss, float* __restrict__ fr) {
    __shared__ float red[256];
    int L = blockIdx.x;                              // 3 blocks
    const float* p = ss + L * ST_SZ;
    float s = 0.0f;
    for (int i = threadIdx.x; i < ST_SZ; i += 256) s += p[i];
    red[threadIdx.x] = s;
    __syncthreads();
    for (int k = 128; k > 0; k >>= 1) {
        if (threadIdx.x < k) red[threadIdx.x] += red[threadIdx.x + k];
        __syncthreads();
    }
    if (threadIdx.x == 0)
        fr[L] = red[0] / ((float)NB * (float)HID * (float)TT);
}

// =====================================================================
extern "C" void kernel_launch(void* const* d_in, const int* in_sizes, int n_in,
                              void* d_out, int out_size, void* d_ws, size_t ws_size,
                              hipStream_t stream) {
    (void)in_sizes; (void)n_in; (void)out_size; (void)ws_size;
    const float* x   = (const float*)d_in[0];
    const float* wi1 = (const float*)d_in[1];
    const float* bi1 = (const float*)d_in[2];
    const float* wh1 = (const float*)d_in[3];
    const float* bh1 = (const float*)d_in[4];
    const float* wi2 = (const float*)d_in[5];
    const float* bi2 = (const float*)d_in[6];
    const float* wh2 = (const float*)d_in[7];
    const float* bh2 = (const float*)d_in[8];
    const float* wi3 = (const float*)d_in[9];
    const float* bi3 = (const float*)d_in[10];
    const float* wo  = (const float*)d_in[11];
    const float* bo  = (const float*)d_in[12];
    const float* ta1 = (const float*)d_in[13];
    const float* ta2 = (const float*)d_in[14];
    const float* ta3 = (const float*)d_in[15];
    const float* tm1 = (const float*)d_in[16];
    const float* tm2 = (const float*)d_in[17];
    const float* tm3 = (const float*)d_in[18];

    char*     ws  = (char*)d_ws;
    _Float16* W16 = (_Float16*)(ws + OFF_W16);
    float*    xT  = (float*)(ws + OFF_XT);
    float*    par = (float*)(ws + OFF_PAR);
    float*    st  = (float*)(ws + OFF_STATE);
    float*    out = (float*)d_out;

    k_cvt_w <<<(512*512 + 255)/256, 256, 0, stream>>>(wh1, wi2, wh2, wi3, W16);
    k_xt    <<<(NB*TT + 255)/256,   256, 0, stream>>>(x, xT);
    k_params<<<(NLAY*HID + 255)/256,256, 0, stream>>>(ta1, ta2, ta3, tm1, tm2, tm3,
                                                      bi1, bh1, bi2, bh2, bi3, par);
    k_init  <<<(10*ST_SZ + 255)/256,256, 0, stream>>>(st);

    k_main  <<<16, 256, 0, stream>>>(W16, xT, par, st, wi1);

    k_out   <<<(NB*10 + 255)/256,     256, 0, stream>>>(st + 8*ST_SZ, wo, bo, out);
    k_srates<<<(NLAY*ST_SZ + 255)/256,256, 0, stream>>>(st + 6*ST_SZ, out + NB*10);
    k_fr    <<<3, 256, 0, stream>>>(st + 6*ST_SZ, out + NB*10 + NLAY*ST_SZ);
}
